// CMC_36069135352483
// MI455X (gfx1250) — matrix operand order repaired and checked by reading, not compiled, not run on hardware
//
#include <hip/hip_runtime.h>

typedef __attribute__((ext_vector_type(16))) _Float16 v16h;
typedef __attribute__((ext_vector_type(8)))  _Float16 v8h;
typedef __attribute__((ext_vector_type(2)))  _Float16 v2h;
typedef __attribute__((ext_vector_type(8)))  float    v8f;
typedef __attribute__((ext_vector_type(2)))  float    v2f;

#define B_   2
#define N_   65536
#define C_   512
#define H_   256
#define W_   256
#define L_   64
#define HUB_ 512
#define HW_  (H_ * W_)

// ---------------------------------------------------------------------------
// One-shot weight conversion f32 -> f16 into workspace (W1h | W2h).
// ---------------------------------------------------------------------------
__global__ void CMC_wcvt(const float* __restrict__ W1, const float* __restrict__ W2,
                         _Float16* __restrict__ out) {
  int i = blockIdx.x * 256 + threadIdx.x;
  const int n1  = HUB_ * L_;
  const int tot = n1 + C_ * HUB_;
  if (i < n1)        out[i] = (_Float16)W1[i];
  else if (i < tot)  out[i] = (_Float16)W2[i - n1];
}

// ---------------------------------------------------------------------------
// img_feat [B,C,H,W] f32 -> imgT [B,H*W,C] f16 (tiled 32x32 LDS transpose).
// Turns the per-point bilinear gather from 1 cacheline/channel into 4
// contiguous 1KB streams per point.
// ---------------------------------------------------------------------------
__global__ __launch_bounds__(256) void CMC_timg(const float* __restrict__ img,
                                                _Float16* __restrict__ outT) {
  __shared__ _Float16 tile[32][34];
  const int p0 = blockIdx.x * 32;   // pixel tile (H*W / 32 = 2048)
  const int c0 = blockIdx.y * 32;   // channel tile (C / 32 = 16)
  const int b  = blockIdx.z;
  const int tx = threadIdx.x;       // 0..31
  const int ty = threadIdx.y;       // 0..7
  const float* src = img + ((size_t)b * C_ + c0) * (size_t)HW_;
#pragma unroll
  for (int i = 0; i < 4; ++i) {
    const int c = ty + i * 8;
    tile[c][tx] = (_Float16)src[(size_t)c * HW_ + p0 + tx];
  }
  __syncthreads();
  _Float16* dst = outT + ((size_t)b * HW_ + p0) * (size_t)C_ + c0;
#pragma unroll
  for (int i = 0; i < 4; ++i) {
    const int pr = ty + i * 8;
    dst[(size_t)pr * C_ + tx] = tile[tx][pr];
  }
}

// ---------------------------------------------------------------------------
// Fragment builders (CDNA5 16-bit WMMA layouts, wave32).
// ---------------------------------------------------------------------------
__device__ inline v16h lds_a_frag(const _Float16* rowp, int kbase, int half) {
  const v8h lo = *(const v8h*)(rowp + kbase + half * 8);
  const v8h hi = *(const v8h*)(rowp + kbase + 16 + half * 8);
  v16h r;
#pragma unroll
  for (int i = 0; i < 8; ++i) { r[i] = lo[i]; r[i + 8] = hi[i]; }
  return r;
}
__device__ inline v16h glb_b_frag(const _Float16* p) {
  const v8h lo = *(const v8h*)p;
  const v8h hi = *(const v8h*)(p + 16);
  v16h r;
#pragma unroll
  for (int i = 0; i < 8; ++i) { r[i] = lo[i]; r[i + 8] = hi[i]; }
  return r;
}

// ---------------------------------------------------------------------------
// Fused kernel: one block = 32 points, 256 threads = 8 waves.
// Each wave owns 4 N-tiles x 2 M-subtiles (B fragment reused across M).
// ---------------------------------------------------------------------------
__global__ __launch_bounds__(256) void CMC_fuse(
    const float* __restrict__ img,  const _Float16* __restrict__ imgT,
    const float* __restrict__ pts,  const float* __restrict__ lidar,
    const float* __restrict__ intr, const float* __restrict__ extr,
    const float* __restrict__ b1,   const float* __restrict__ b2,
    const _Float16* __restrict__ W1h, const _Float16* __restrict__ W2h,
    float* __restrict__ out, const int useT)
{
  __shared__ _Float16 sA[32][72];    // lidar tile f16 (padded rows)
  __shared__ _Float16 sH[32][520];   // hidden f16
  __shared__ _Float16 sM[32][520];   // mapped f16
  __shared__ float    sWgt[32][4];
  __shared__ int      sOff[32][4];
  __shared__ int      sIns[32];

  const int tid = threadIdx.x;
  const int b   = blockIdx.x >> 11;           // 2048 tiles per batch
  const int n0  = (blockIdx.x & 2047) << 5;   // first point of 32-point tile

  // ---- stage 0: projection + bilinear setup (threads 0..31)
  if (tid < 32) {
    const int p = tid;
    const float* P = pts + ((size_t)(b * N_ + n0 + p)) * 3;
    const float x = P[0], y = P[1], z = P[2];
    const float* E = extr + b * 16;
    const float cxp = E[0]*x + E[1]*y + E[2]*z  + E[3];
    const float cyp = E[4]*x + E[5]*y + E[6]*z  + E[7];
    const float czp = E[8]*x + E[9]*y + E[10]*z + E[11];
    const float* Km = intr + b * 9;
    const float pr0 = Km[0]*cxp + Km[1]*cyp + Km[2]*czp;
    const float pr1 = Km[3]*cxp + Km[4]*cyp + Km[5]*czp;
    const float pr2 = Km[6]*cxp + Km[7]*cyp + Km[8]*czp;
    const float u = pr0 / pr2, v = pr1 / pr2;
    const bool inside = (u >= 0.f) && (u < (float)W_) &&
                        (v >= 0.f) && (v < (float)H_) && (czp > 0.f);
    sIns[p] = inside ? 1 : 0;
    if (inside) {
      const float x0f = floorf(u), y0f = floorf(v);
      const float wx = u - x0f,  wy = v - y0f;
      const int x0 = (int)x0f, y0 = (int)y0f;
      const bool vx1 = (x0 + 1) < W_, vy1 = (y0 + 1) < H_;
      const int x1 = vx1 ? x0 + 1 : W_ - 1;
      const int y1 = vy1 ? y0 + 1 : H_ - 1;
      sWgt[p][0] = (1.f - wx) * (1.f - wy);
      sWgt[p][1] = vx1 ? wx * (1.f - wy) : 0.f;
      sWgt[p][2] = vy1 ? (1.f - wx) * wy : 0.f;
      sWgt[p][3] = (vx1 && vy1) ? wx * wy : 0.f;
      sOff[p][0] = y0 * W_ + x0;
      sOff[p][1] = y0 * W_ + x1;
      sOff[p][2] = y1 * W_ + x0;
      sOff[p][3] = y1 * W_ + x1;
    }
  }
  // ---- stage lidar tile -> LDS f16 (coalesced)
  for (int idx = tid; idx < 32 * L_; idx += 256) {
    const int p = idx >> 6, l = idx & 63;
    sA[p][l] = (_Float16)lidar[((size_t)(b * N_ + n0 + p)) * L_ + l];
  }
  __syncthreads();

  int cnt = 0;
#pragma unroll
  for (int i = 0; i < 32; ++i) cnt += sIns[i];

  const int wid  = tid >> 5;
  const int lane = tid & 31;
  const int half = lane >> 4;
  const int row  = lane & 15;

  if (cnt < 32) {   // block-uniform (essentially always taken)
    // ---- stage 1: hidden = relu(lidar @ W1^T + b1)
    v16h a0[2], a1[2];
#pragma unroll
    for (int m = 0; m < 2; ++m) {
      a0[m] = lds_a_frag(&sA[m * 16 + row][0], 0,  half);
      a1[m] = lds_a_frag(&sA[m * 16 + row][0], 32, half);
    }
#pragma unroll
    for (int t = 0; t < 4; ++t) {
      const int col = (wid * 4 + t) * 16 + row;
      const v16h bb0 = glb_b_frag(W1h + (size_t)col * L_ + half * 8);
      const v16h bb1 = glb_b_frag(W1h + (size_t)col * L_ + 32 + half * 8);
      const float bias = b1[col];
#pragma unroll
      for (int m = 0; m < 2; ++m) {          // B fragment reused across M
        v8f c = {};
        c = __builtin_amdgcn_wmma_f32_16x16x32_f16(false, a0[m], false, bb0, (short)0, c, false, false);
        c = __builtin_amdgcn_wmma_f32_16x16x32_f16(false, a1[m], false, bb1, (short)0, c, false, false);
#pragma unroll
        for (int r = 0; r < 8; ++r)
          sH[m * 16 + r + 8 * half][col] = (_Float16)fmaxf(c[r] + bias, 0.f);
      }
    }
    __syncthreads();

    // ---- stage 2: mapped = hidden @ W2^T + b2
    v8f acc[2][4] = {};
    for (int k0 = 0; k0 < 16; ++k0) {
      const v16h ak0 = lds_a_frag(&sH[row][0],      k0 * 32, half);
      const v16h ak1 = lds_a_frag(&sH[16 + row][0], k0 * 32, half);
#pragma unroll
      for (int t = 0; t < 4; ++t) {
        const int col = (wid * 4 + t) * 16 + row;
        const v16h bk = glb_b_frag(W2h + (size_t)col * HUB_ + k0 * 32 + half * 8);
        acc[0][t] = __builtin_amdgcn_wmma_f32_16x16x32_f16(false, ak0, false, bk, (short)0, acc[0][t], false, false);
        acc[1][t] = __builtin_amdgcn_wmma_f32_16x16x32_f16(false, ak1, false, bk, (short)0, acc[1][t], false, false);
      }
    }
#pragma unroll
    for (int t = 0; t < 4; ++t) {
      const int col = (wid * 4 + t) * 16 + row;
      const float bias = b2[col];
#pragma unroll
      for (int m = 0; m < 2; ++m)
#pragma unroll
        for (int r = 0; r < 8; ++r)
          sM[m * 16 + r + 8 * half][col] = (_Float16)(acc[m][t][r] + bias);
    }
  }
  __syncthreads();

  // ---- final: per point select (bilinear | MLP), thread = channel pair
  const size_t outBase = ((size_t)b * N_ + n0) * (size_t)C_;
  const int c2 = tid * 2;
  for (int p = 0; p < 32; ++p) {
    const size_t ob = outBase + (size_t)p * C_;
    v2f val;
    if (sIns[p]) {
      const float w0 = sWgt[p][0], w1 = sWgt[p][1], w2 = sWgt[p][2], w3 = sWgt[p][3];
      const int   o0 = sOff[p][0], o1 = sOff[p][1], o2 = sOff[p][2], o3 = sOff[p][3];
      if (useT) {   // transposed path: 4 contiguous 1KB streams per point
        const _Float16* tb = imgT + (size_t)b * HW_ * (size_t)C_;
        const v2h q0 = *(const v2h*)(tb + (size_t)o0 * C_ + c2);
        const v2h q1 = *(const v2h*)(tb + (size_t)o1 * C_ + c2);
        const v2h q2 = *(const v2h*)(tb + (size_t)o2 * C_ + c2);
        const v2h q3 = *(const v2h*)(tb + (size_t)o3 * C_ + c2);
        val[0] = w0*(float)q0[0] + w1*(float)q1[0] + w2*(float)q2[0] + w3*(float)q3[0];
        val[1] = w0*(float)q0[1] + w1*(float)q1[1] + w2*(float)q2[1] + w3*(float)q3[1];
      } else {      // fallback: channel-major gather
        const float* ic0 = img + ((size_t)(b * C_ + c2)) * (size_t)HW_;
        const float* ic1 = ic0 + HW_;
        val[0] = w0*ic0[o0] + w1*ic0[o1] + w2*ic0[o2] + w3*ic0[o3];
        val[1] = w0*ic1[o0] + w1*ic1[o1] + w2*ic1[o2] + w3*ic1[o3];
      }
    } else {
      val[0] = (float)sM[p][c2];
      val[1] = (float)sM[p][c2 + 1];
    }
    *(v2f*)(out + ob + c2) = val;   // 8B/lane, 256B/wave coalesced
  }
}

extern "C" void kernel_launch(void* const* d_in, const int* in_sizes, int n_in,
                              void* d_out, int out_size, void* d_ws, size_t ws_size,
                              hipStream_t stream) {
  (void)in_sizes; (void)n_in; (void)out_size;
  const float* img   = (const float*)d_in[0];
  const float* pts   = (const float*)d_in[1];
  const float* lidar = (const float*)d_in[2];
  const float* intr  = (const float*)d_in[3];
  const float* extr  = (const float*)d_in[4];
  const float* W1    = (const float*)d_in[5];
  const float* b1    = (const float*)d_in[6];
  const float* W2    = (const float*)d_in[7];
  const float* b2    = (const float*)d_in[8];
  float* out = (float*)d_out;

  _Float16* wh = (_Float16*)d_ws;
  const size_t wbytes   = (size_t)(HUB_ * L_ + C_ * HUB_) * sizeof(_Float16);
  const size_t imgT_off = (wbytes + 255) & ~(size_t)255;
  const size_t imgT_sz  = (size_t)B_ * HW_ * C_ * sizeof(_Float16);   // 134 MB
  const int useT        = (ws_size >= imgT_off + imgT_sz) ? 1 : 0;
  _Float16* imgT = (_Float16*)((char*)d_ws + imgT_off);

  const int tot = HUB_ * L_ + C_ * HUB_;
  CMC_wcvt<<<(tot + 255) / 256, 256, 0, stream>>>(W1, W2, wh);
  if (useT)
    CMC_timg<<<dim3(HW_ / 32, C_ / 32, B_), dim3(32, 8), 0, stream>>>(img, imgT);

  CMC_fuse<<<dim3((B_ * N_) / 32), 256, 0, stream>>>(
      img, imgT, pts, lidar, intr, extr, b1, b2, wh, wh + HUB_ * L_, out, useT);
}
